// LigerFusedLinearCrossEntropyLoss_48275432407510
// MI455X (gfx1250) — compile-verified, matrix operand order checked by reading, not run on hardware
//
#include <hip/hip_runtime.h>

// ---------------------------------------------------------------------------
// Fused linear + cross-entropy (Liger-style), BETA=0 -> pure CE.
//   logits = X @ W^T + bias ;  loss = mean over valid tokens of
//   logsumexp(logits[t,:]) - logits[t, target[t]]
// Strategy: never materialize logits; online softmax from WMMA accumulators.
// HBM-bound on W (512 MB) -> single streaming pass, bf16 WMMA f32-accum.
// ---------------------------------------------------------------------------

#define N_TOK   4096
#define HIDDEN  4096
#define VOCAB   32000
#define IGNORE_INDEX (-100)

#define TOKS        16                    // tokens per workgroup
#define NV_SPLITS   25                    // vocab partitions (grid.y)
#define VCHUNK      (VOCAB / NV_SPLITS)   // 1280 vocab rows per split
#define COLS_PER_WG 128                   // 8 waves x 16 columns per pass
#define KCHUNK      1024                  // K staging chunk for A in LDS
#define LDS_KP      (KCHUNK + 8)          // +8 bf16 pad -> conflict-free rows

typedef __bf16 bf16;
typedef __attribute__((ext_vector_type(4)))  __bf16 v4bf;
typedef __attribute__((ext_vector_type(8)))  __bf16 v8bf;
typedef __attribute__((ext_vector_type(16))) __bf16 v16bf;
typedef __attribute__((ext_vector_type(8)))  float  v8f;

// ---------------------------------------------------------------------------
// Kernel 1: per (token-block, vocab-split) -> per-token (max, sumexp) partials
// plus the target logit. 8 waves, each does a 16x16 WMMA tile per pass.
// ---------------------------------------------------------------------------
__global__ __launch_bounds__(256) void flce_gemm_partial(
    const float* __restrict__ W,      // [VOCAB][HIDDEN]
    const float* __restrict__ X,      // [N_TOK][HIDDEN]
    const int*   __restrict__ tgt,    // [N_TOK]
    const float* __restrict__ bias,   // [VOCAB]
    float* __restrict__ part_m,       // [N_TOK][NV_SPLITS]
    float* __restrict__ part_s,       // [N_TOK][NV_SPLITS]
    float* __restrict__ tgt_logit)    // [N_TOK]
{
    __shared__ bf16  lA[TOKS * LDS_KP];   // ~33 KB bf16 A chunk
    __shared__ float red_m[256 * 8];      // per-lane accumulator dump
    __shared__ float red_s[256 * 8];

    const int tid  = threadIdx.x;
    const int wave = tid >> 5;
    const int lane = tid & 31;
    const int hi   = lane >> 4;           // half of wave
    const int l16  = lane & 15;
    const int bk   = hi << 3;             // base K within 32-step: 0 or 8

    const int tok0    = blockIdx.x * TOKS;
    const int v_begin = blockIdx.y * VCHUNK;

    // Targets for the 8 token rows this lane's C fragment covers.
    int tgt_l[8];
#pragma unroll
    for (int j = 0; j < 8; ++j) tgt_l[j] = tgt[tok0 + (hi << 3) + j];

    // Online-softmax state: 8 (token) slots per lane.
    float m_acc[8], s_acc[8];
#pragma unroll
    for (int j = 0; j < 8; ++j) { m_acc[j] = -1e30f; s_acc[j] = 0.f; }

    for (int v0 = v_begin; v0 < v_begin + VCHUNK; v0 += COLS_PER_WG) {
        const int    n_g = v0 + (wave << 4) + l16;   // this lane's vocab row
        const float* wr  = W + (size_t)n_g * HIDDEN;
        v8f c = {0.f, 0.f, 0.f, 0.f, 0.f, 0.f, 0.f, 0.f};

        for (int kc = 0; kc < HIDDEN; kc += KCHUNK) {
            // ---- stage A chunk (16 x KCHUNK) into LDS as bf16 ----
            __syncthreads();
            for (int i = tid * 4; i < TOKS * KCHUNK; i += 256 * 4) {
                const int row = i / KCHUNK;
                const int col = i - row * KCHUNK;
                const float4 f =
                    *(const float4*)(X + (size_t)(tok0 + row) * HIDDEN + kc + col);
                v4bf p;
                p[0] = (bf16)f.x; p[1] = (bf16)f.y;
                p[2] = (bf16)f.z; p[3] = (bf16)f.w;
                *(v4bf*)(lA + row * LDS_KP + col) = p;
            }
            __syncthreads();

            // ---- K loop: 32 per WMMA ----
#pragma unroll 4
            for (int k0 = 0; k0 < KCHUNK; k0 += 32) {
                // A fragment: row M=l16; lanes 0-15 K={0..7,16..23},
                // lanes 16-31 K={8..15,24..31} (ISA 16-bit A layout).
                const bf16* ap   = lA + l16 * LDS_KP + k0 + bk;
                const v8bf  a_lo = *(const v8bf*)ap;
                const v8bf  a_hi = *(const v8bf*)(ap + 16);
                const v16bf a    = __builtin_shufflevector(
                    a_lo, a_hi, 0,1,2,3,4,5,6,7,8,9,10,11,12,13,14,15);

                // B fragment: column N=l16 from W row n_g, same K split.
                const float* bp = wr + kc + k0 + bk;
                const float4 b0 = *(const float4*)(bp);
                const float4 b1 = *(const float4*)(bp + 4);
                const float4 b2 = *(const float4*)(bp + 16);
                const float4 b3 = *(const float4*)(bp + 20);
                v16bf b;
                b[0]  = (bf16)b0.x; b[1]  = (bf16)b0.y;
                b[2]  = (bf16)b0.z; b[3]  = (bf16)b0.w;
                b[4]  = (bf16)b1.x; b[5]  = (bf16)b1.y;
                b[6]  = (bf16)b1.z; b[7]  = (bf16)b1.w;
                b[8]  = (bf16)b2.x; b[9]  = (bf16)b2.y;
                b[10] = (bf16)b2.z; b[11] = (bf16)b2.w;
                b[12] = (bf16)b3.x; b[13] = (bf16)b3.y;
                b[14] = (bf16)b3.z; b[15] = (bf16)b3.w;

                c = __builtin_amdgcn_wmma_f32_16x16x32_bf16(
                        false, a, false, b, (short)0, c, false, false);
            }
        }

        // ---- fold this pass's 16x16 logits into online softmax state ----
        const float bn = bias[n_g];
#pragma unroll
        for (int j = 0; j < 8; ++j) {
            const float l  = c[j] + bn;               // logit(token, n_g)
            const float mo = m_acc[j];
            const float mn = fmaxf(mo, l);
            s_acc[j] = s_acc[j] * __expf(mo - mn) + __expf(l - mn);
            m_acc[j] = mn;
            if (tgt_l[j] == n_g)                      // single writer globally
                tgt_logit[tok0 + (hi << 3) + j] = l;
        }
    }

    // ---- combine 8 waves x 16 lanes of per-token partials via LDS ----
    const int base = tid * 8;
#pragma unroll
    for (int j = 0; j < 8; ++j) { red_m[base + j] = m_acc[j];
                                  red_s[base + j] = s_acc[j]; }
    __syncthreads();

    if (tid < TOKS) {
        const int slot   = tid & 7;    // accumulator slot for this token
        const int half_t = tid >> 3;   // which lane half held it
        float M = -1e30f, S = 0.f;
        for (int w = 0; w < 8; ++w) {
            for (int l = 0; l < 16; ++l) {
                const int   idx = ((w * 32) + half_t * 16 + l) * 8 + slot;
                const float m = red_m[idx], s = red_s[idx];
                const float Mn = fmaxf(M, m);
                S = S * __expf(M - Mn) + s * __expf(m - Mn);
                M = Mn;
            }
        }
        part_m[(size_t)(tok0 + tid) * NV_SPLITS + blockIdx.y] = M;
        part_s[(size_t)(tok0 + tid) * NV_SPLITS + blockIdx.y] = S;
    }
}

// ---------------------------------------------------------------------------
// Kernel 2: combine vocab-split partials per token -> scalar mean loss.
// ---------------------------------------------------------------------------
__global__ __launch_bounds__(256) void flce_reduce(
    const float* __restrict__ part_m, const float* __restrict__ part_s,
    const float* __restrict__ tgt_logit, const int* __restrict__ tgt,
    float* __restrict__ out)
{
    __shared__ float r_sum[256];
    __shared__ float r_cnt[256];

    float sum = 0.f, cnt = 0.f;
    for (int t = threadIdx.x; t < N_TOK; t += 256) {
        if (tgt[t] == IGNORE_INDEX) continue;
        const float* pm = part_m + (size_t)t * NV_SPLITS;
        const float* ps = part_s + (size_t)t * NV_SPLITS;
        float M = -1e30f, S = 0.f;
        for (int j = 0; j < NV_SPLITS; ++j) {
            const float m = pm[j], s = ps[j];
            const float Mn = fmaxf(M, m);
            S = S * __expf(M - Mn) + s * __expf(m - Mn);
            M = Mn;
        }
        sum += (M + __logf(S)) - tgt_logit[t];   // logZ - target logit
        cnt += 1.f;
    }
    r_sum[threadIdx.x] = sum;
    r_cnt[threadIdx.x] = cnt;
    __syncthreads();
    for (int off = 128; off > 0; off >>= 1) {
        if (threadIdx.x < off) {
            r_sum[threadIdx.x] += r_sum[threadIdx.x + off];
            r_cnt[threadIdx.x] += r_cnt[threadIdx.x + off];
        }
        __syncthreads();
    }
    if (threadIdx.x == 0) out[0] = r_sum[0] / fmaxf(r_cnt[0], 1.f);
}

// ---------------------------------------------------------------------------
extern "C" void kernel_launch(void* const* d_in, const int* in_sizes, int n_in,
                              void* d_out, int out_size, void* d_ws, size_t ws_size,
                              hipStream_t stream) {
    const float* W    = (const float*)d_in[0];   // lin_weight [VOCAB*HIDDEN]
    const float* X    = (const float*)d_in[1];   // _input     [N_TOK*HIDDEN]
    const int*   tgt  = (const int*)  d_in[2];   // target     [N_TOK]
    // d_in[3] ref_hidden, d_in[4] ref_weight: BETA == 0 -> unused.
    const float* bias = (const float*)d_in[5];   // bias       [VOCAB]

    float* part_m    = (float*)d_ws;
    float* part_s    = part_m + (size_t)N_TOK * NV_SPLITS;
    float* tgt_logit = part_s + (size_t)N_TOK * NV_SPLITS;
    (void)ws_size; (void)in_sizes; (void)n_in; (void)out_size;

    // grid.x fast (token blocks) so co-resident WGs share one W slice in L2.
    dim3 grid(N_TOK / TOKS, NV_SPLITS);
    flce_gemm_partial<<<grid, 256, 0, stream>>>(W, X, tgt, bias,
                                                part_m, part_s, tgt_logit);
    flce_reduce<<<1, 256, 0, stream>>>(part_m, part_s, tgt_logit, tgt,
                                       (float*)d_out);
}
